// Decoder_36344013258863
// MI455X (gfx1250) — matrix operand order repaired and checked by reading, not compiled, not run on hardware
//
#include <hip/hip_runtime.h>
#include <hip/hip_bf16.h>
#include <math.h>

typedef __attribute__((ext_vector_type(16))) __bf16 v16bf;
typedef __attribute__((ext_vector_type(8)))  __bf16 v8bf;
typedef __attribute__((ext_vector_type(8)))  float  v8f;

#define BLOCK_THREADS 512
#define EDGES_PER_BLOCK 256    // 16 waves * 16 edges

// ---------------------------------------------------------------------------
// Kernel 1: pack W1 [128 x 256] f32 into bf16 WMMA B-fragment-linear layout:
//   fragment (nt, kt) -> 32 lanes -> 16 contiguous bf16 per lane
//   lane l (l<16):  column n = nt*16 + l,      K = kt*32 + 0..15
//   lane l (l>=16): column n = nt*16 + (l-16), K = kt*32 + 16..31
// Total: 64 fragments * 32 lanes * 16 elems = 32768 bf16 = 64 KB
// ---------------------------------------------------------------------------
__global__ void w1_pack_kernel(const float* __restrict__ W1,
                               __bf16* __restrict__ w1f) {
    int tid = blockIdx.x * blockDim.x + threadIdx.x;
    if (tid >= 64 * 32 * 16) return;
    int i    = tid & 15;
    int lane = (tid >> 4) & 31;
    int frag = tid >> 9;            // 0..63
    int nt   = frag >> 3;           // 0..7  (N tile)
    int kt   = frag & 7;            // 0..7  (K tile)
    int n = nt * 16 + (lane & 15);
    int k = kt * 32 + ((lane & 16) ? 16 : 0) + i;
    w1f[tid] = (__bf16)W1[n * 256 + k];
}

// ---------------------------------------------------------------------------
// Kernel 1b: convert node features x [N,128] f32 -> bf16 (row-major).
// Each thread converts 8 contiguous elements.
// ---------------------------------------------------------------------------
__global__ void x_bf16_kernel(const float* __restrict__ x,
                              __bf16* __restrict__ xb, int total) {
    int base = (blockIdx.x * blockDim.x + threadIdx.x) * 8;
    if (base >= total) return;
    v8bf v;
#pragma unroll
    for (int i = 0; i < 8; ++i) v[i] = (__bf16)x[base + i];
    *(v8bf*)(xb + base) = v;
}

// Pack 8+8 contiguous floats into a bf16 fragment register (f32 source path).
__device__ inline v16bf pack16_f32(const float* __restrict__ p0,
                                   const float* __restrict__ p1) {
    v16bf v;
#pragma unroll
    for (int i = 0; i < 8; ++i) v[i] = (__bf16)p0[i];
#pragma unroll
    for (int i = 0; i < 8; ++i) v[8 + i] = (__bf16)p1[i];
    return v;
}

// Pack 8+8 contiguous bf16 (two 16-byte loads, no conversion).
__device__ inline v16bf pack16_bf(const __bf16* __restrict__ p0,
                                  const __bf16* __restrict__ p1) {
    v8bf lo = *(const v8bf*)p0;
    v8bf hi = *(const v8bf*)p1;
    v16bf v;
#pragma unroll
    for (int i = 0; i < 8; ++i) { v[i] = lo[i]; v[8 + i] = hi[i]; }
    return v;
}

// ---------------------------------------------------------------------------
// Kernel 2: fused edge-MLP.
//   Block = 512 threads = 16 waves; each wave handles 16 edges (M=16),
//   so 256 edges/block — amortizes the 64 KB W1 LDS staging over 2x the
//   gather bytes (0.75 KB/edge of L2 reads vs 1.0 KB at 128 edges/block).
//   Wave: gather concat(x[src],x[dst]) as 8 bf16 A-fragments (K=256),
//   8 N-tiles x 8 K-tiles of v_wmma_f32_16x16x32_bf16 vs LDS-resident W1,
//   fused bias+ReLU+W2-dot, shfl reduce, bias+ReLU+sigmoid, store.
//   XB selects bf16-preconverted gather (half the L2 bytes, no cvt in loop).
// ---------------------------------------------------------------------------
template <bool XB>
__launch_bounds__(BLOCK_THREADS)
__global__ void edge_mlp_kernel(const float*  __restrict__ x,
                                const __bf16* __restrict__ xb,
                                const int*    __restrict__ ei,   // [2, E]
                                const __bf16* __restrict__ w1f,  // packed 64 KB
                                const float*  __restrict__ b1,
                                const float*  __restrict__ W2,
                                const float*  __restrict__ b2,
                                float* __restrict__ out,
                                int E) {
    __shared__ v16bf w1lds[64 * 32];   // 64 KB: 64 fragments x 32 lanes

    // Cooperative stage of W1 fragments into LDS (4096 x 16B).
    {
        const uint4* __restrict__ src = (const uint4*)w1f;
        uint4* dst = (uint4*)w1lds;
#pragma unroll
        for (int it = 0; it < 4096 / BLOCK_THREADS; ++it) {
            int i = it * BLOCK_THREADS + threadIdx.x;
            dst[i] = src[i];
        }
    }
    __syncthreads();

    const int wave = threadIdx.x >> 5;     // 0..15
    const int lane = threadIdx.x & 31;
    const int hi   = (lane >> 4) & 1;      // lane half
    long long e0 = (long long)blockIdx.x * EDGES_PER_BLOCK + wave * 16;
    if (e0 >= E) return;                   // uniform per-wave exit (after sync)

    // Edge endpoints for this wave's 16 edges (row r = lane & 15).
    int r = lane & 15;
    long long eg = e0 + r;
    if (eg >= E) eg = E - 1;               // clamp tail (stores are guarded)
    const int nsrc = ei[eg];
    const int ndst = ei[(long long)E + eg];

    // Gather A-fragments: 16 edges x K=256 (K 0..127 = src, 128..255 = dst).
    // A k order used here (contiguous, same as packed W1): lane<16 -> K = kt*32 + 0..15,
    //                           lane>=16 -> +16.
    v16bf a[8];
#pragma unroll
    for (int kt = 0; kt < 8; ++kt) {
        const int node = (kt < 4) ? nsrc : ndst;
        const int koff = (kt & 3) * 32 + hi * 16;
        if (XB) {
            const __bf16* p = xb + (long long)node * 128 + koff;
            a[kt] = pack16_bf(p, p + 8);
        } else {
            const float* p = x + (long long)node * 128 + koff;
            a[kt] = pack16_f32(p, p + 8);
        }
    }

    // Accumulate layer-2 partial dot products per C-fragment slot.
    float partial[8];
#pragma unroll
    for (int v = 0; v < 8; ++v) partial[v] = 0.f;

    const int ncol = lane & 15;            // C layout: N = lane & 15
#pragma unroll
    for (int nt = 0; nt < 8; ++nt) {
        v8f acc = {};
#pragma unroll
        for (int kt = 0; kt < 8; ++kt) {
            v16bf b = w1lds[(nt * 8 + kt) * 32 + lane];
            acc = __builtin_amdgcn_wmma_f32_16x16x32_bf16(
                false, a[kt], false, b, (short)0, acc, false, false);
        }
        const float b1v = b1[nt * 16 + ncol];
        const float w2v = W2[nt * 16 + ncol];
#pragma unroll
        for (int v = 0; v < 8; ++v) {      // C slot v: edge M = v + 8*hi
            float h = acc[v] + b1v;
            h = h > 0.f ? h : 0.f;         // ReLU (layer 1)
            partial[v] += h * w2v;         // layer-2 dot contribution
        }
    }

    // Reduce over N (16 lanes per half; xor masks stay inside each half).
#pragma unroll
    for (int v = 0; v < 8; ++v) {
        float s = partial[v];
        s += __shfl_xor(s, 1);
        s += __shfl_xor(s, 2);
        s += __shfl_xor(s, 4);
        s += __shfl_xor(s, 8);
        partial[v] = s;
    }

    // Lanes 0 and 16 hold full sums for edges e0+v and e0+8+v.
    if ((lane & 15) == 0) {
        const float bb2 = b2[0];
        const int mbase = hi * 8;
#pragma unroll
        for (int v = 0; v < 8; ++v) {
            long long e = e0 + mbase + v;
            if (e < E) {
                float h = partial[v] + bb2;
                h = h > 0.f ? h : 0.f;                 // ReLU (layer 2)
                out[e] = 1.f / (1.f + expf(-h));       // sigmoid
            }
        }
    }
}

// ---------------------------------------------------------------------------
extern "C" void kernel_launch(void* const* d_in, const int* in_sizes, int n_in,
                              void* d_out, int out_size, void* d_ws, size_t ws_size,
                              hipStream_t stream) {
    const float* x  = (const float*)d_in[0];
    const int*   ei = (const int*)  d_in[1];   // edge_index, [2, E]
    const float* W1 = (const float*)d_in[2];   // [128, 256]
    const float* b1 = (const float*)d_in[3];   // [128]
    const float* W2 = (const float*)d_in[4];   // [1, 128]
    const float* b2 = (const float*)d_in[5];   // [1]
    float* out = (float*)d_out;

    const int E      = in_sizes[1] / 2;
    const int xtotal = in_sizes[0];            // N_NODES * 128

    __bf16* w1f = (__bf16*)d_ws;               // 64 KB packed W1 at ws offset 0
    w1_pack_kernel<<<128, 256, 0, stream>>>(W1, w1f);

    const size_t need_xb = (size_t)65536 + (size_t)xtotal * sizeof(__bf16);
    const int blocks = (E + EDGES_PER_BLOCK - 1) / EDGES_PER_BLOCK;

    if (ws_size >= need_xb) {
        // Pre-convert x to bf16: halves gather bytes, removes cvt from loop.
        __bf16* xb = (__bf16*)((char*)d_ws + 65536);
        x_bf16_kernel<<<(xtotal / 8 + 255) / 256, 256, 0, stream>>>(x, xb, xtotal);
        edge_mlp_kernel<true><<<blocks, BLOCK_THREADS, 0, stream>>>(
            x, xb, ei, w1f, b1, W2, b2, out, E);
    } else {
        edge_mlp_kernel<false><<<blocks, BLOCK_THREADS, 0, stream>>>(
            x, (const __bf16*)nullptr, ei, w1f, b1, W2, b2, out, E);
    }
}
